// DynamicCompression_87754771792448
// MI455X (gfx1250) — matrix operand order repaired and checked by reading, not compiled, not run on hardware
//
#include <hip/hip_runtime.h>
#include <hip/hip_bf16.h>

typedef __attribute__((ext_vector_type(16))) _Float16 v16h;
typedef __attribute__((ext_vector_type(8)))  float    v8f;

#define B_    4
#define S_    8192
#define D_    1024
#define H_    512
#define Q_    256
#define KSEL  4096

// ---------------------------------------------------------------------------
// Pack w1 [D,H] f32 into WMMA-B fragment-major f16 layout:
//   w1p[((ks*32 + nt)*32 + lane)*16 + slot]
//   lane: n = nt*16 + (lane&15), kh = (lane>>4)*8
//   slot s <-> k = ks*32 + (s>>3)*16 + kh + (s&7)   (16-bit B 32x16 ISA striping)
__global__ __launch_bounds__(256) void pack_w1_kernel(const float* __restrict__ w1,
                                                      _Float16* __restrict__ w1p) {
    const int o    = blockIdx.x * 256 + threadIdx.x;   // 0 .. D*H-1
    const int slot = o & 15;
    const int lane = (o >> 4) & 31;
    const int nt   = (o >> 9) & 31;
    const int ks   = o >> 14;
    const int n = nt * 16 + (lane & 15);
    const int k = ks * 32 + ((slot >> 3) << 4) + (lane >> 4) * 8 + (slot & 7);
    w1p[o] = (_Float16)w1[(size_t)k * H_ + n];
}

// ------------------------------------------------- importance scorer (WMMA)
// One block = 32 tokens (2 M-tiles), 256 threads (8 waves), each wave owns
// 4 N-tiles for BOTH M-tiles (B fragment reused for 2 WMMAs).
__global__ __launch_bounds__(256) void scorer_kernel(
    const float* __restrict__ x, const _Float16* __restrict__ w1p,
    const float* __restrict__ b1, const float* __restrict__ ln_g,
    const float* __restrict__ ln_b, const float* __restrict__ w2,
    const float* __restrict__ b2, float* __restrict__ scores) {

    // 64 KB: A fragments packed (2 x 32KB), later reused as f32 h-tile 32x512
    __shared__ _Float16 smem[32 * D_];
    float* hbuf = (float*)smem;

    const int tid  = threadIdx.x;
    const int lane = tid & 31;
    const int wave = tid >> 5;                 // 0..7
    const int hi   = lane >> 4;
    const int row0 = blockIdx.x * 32;          // token index in flattened [B*S]

    // Stage 32x1024 x-tile into LDS directly in A-fragment layout (f16).
    // A 16x32 striping: element (m, ko): lane = 16*((ko>>3)&1) + m,
    // slot = ((ko>>4)<<3) | (ko&7).
    const float* xbase = x + (size_t)row0 * D_;
    for (int i = tid; i < 32 * D_; i += 256) {
        const int mrow = i >> 10;              // 0..31
        const int k    = i & 1023;
        const int ks   = k >> 5;
        const int ko   = k & 31;
        const int l    = ((ko >> 3) & 1) * 16 + (mrow & 15);
        const int slot = ((ko >> 4) << 3) | (ko & 7);
        const int mt   = mrow >> 4;
        smem[mt * 16384 + (ks * 32 + l) * 16 + slot] = (_Float16)xbase[i];
    }
    __syncthreads();

    v8f acc[2][4] = {};
    for (int ks = 0; ks < D_ / 32; ++ks) {
        const v16h a0 = *(const v16h*)&smem[(ks * 32 + lane) * 16];
        const v16h a1 = *(const v16h*)&smem[16384 + (ks * 32 + lane) * 16];
        if (ks + 1 < D_ / 32)   // pull next K-slab of packed B toward the WGP
            __builtin_prefetch((const void*)(w1p + ((size_t)(ks + 1) * 32 + wave * 4) * 512),
                               0, 1);
#pragma unroll
        for (int t = 0; t < 4; ++t) {
            const int nt = wave * 4 + t;
            const v16h bf = *(const v16h*)&w1p[(((size_t)ks * 32 + nt) * 32 + lane) * 16];
            acc[0][t] = __builtin_amdgcn_wmma_f32_16x16x32_f16(
                false, a0, false, bf, (short)0, acc[0][t], false, false);
            acc[1][t] = __builtin_amdgcn_wmma_f32_16x16x32_f16(
                false, a1, false, bf, (short)0, acc[1][t], false, false);
        }
    }
    __syncthreads();                            // all LDS A reads done

    // Spill D tiles (+b1) to f32 hbuf[32][512]:
    // D layout: VGPR r, lane l -> (m = r + 8*(l>>4), n = l&15)
#pragma unroll
    for (int t = 0; t < 4; ++t) {
        const int n = (wave * 4 + t) * 16 + (lane & 15);
        const float bb = b1[n];
#pragma unroll
        for (int mt = 0; mt < 2; ++mt)
#pragma unroll
            for (int r = 0; r < 8; ++r)
                hbuf[(mt * 16 + r + hi * 8) * H_ + n] = acc[mt][t][r] + bb;
    }
    __syncthreads();

    // LayerNorm + ReLU + dot(w2) + sigmoid: 8 waves, 4 rows each.
    for (int rr = 0; rr < 4; ++rr) {
        const int mm = wave * 4 + rr;
        float s = 0.f, s2 = 0.f, vals[16];
#pragma unroll
        for (int j = 0; j < 16; ++j) {
            const float v = hbuf[mm * H_ + lane + j * 32];
            vals[j] = v; s += v; s2 += v * v;
        }
#pragma unroll
        for (int o = 16; o > 0; o >>= 1) {
            s  += __shfl_xor(s, o, 32);
            s2 += __shfl_xor(s2, o, 32);
        }
        const float mu   = s * (1.0f / H_);
        const float var  = s2 * (1.0f / H_) - mu * mu;
        const float rstd = rsqrtf(var + 1e-5f);
        float dot = 0.f;
#pragma unroll
        for (int j = 0; j < 16; ++j) {
            const int n = lane + j * 32;
            float v = (vals[j] - mu) * rstd * ln_g[n] + ln_b[n];
            v = fmaxf(v, 0.f);
            dot += v * w2[n];
        }
#pragma unroll
        for (int o = 16; o > 0; o >>= 1) dot += __shfl_xor(dot, o, 32);
        if (lane == 0) {
            const float z = dot + b2[0];
            scores[row0 + mm] = 1.f / (1.f + __expf(-z));
        }
    }
}

// ----------------------------------------- exact top-k via stable rank count
__global__ __launch_bounds__(256) void rank_kernel(const float* __restrict__ scores,
                                                   int* __restrict__ idx) {
    __shared__ float tile[256];
    const int b = blockIdx.y;
    const int i = blockIdx.x * 256 + threadIdx.x;
    const float* sb = scores + b * S_;
    const float si = sb[i];
    int count = 0;
    for (int j0 = 0; j0 < S_; j0 += 256) {
        tile[threadIdx.x] = sb[j0 + threadIdx.x];
        __syncthreads();
#pragma unroll 8
        for (int jj = 0; jj < 256; ++jj) {
            const float sj = tile[jj];
            const int   j  = j0 + jj;
            count += (sj > si) || (sj == si && j < i);
        }
        __syncthreads();
    }
    if (count < KSEL) idx[b * KSEL + count] = i;
}

// ---------------------------------------------------------------- zero init
__global__ __launch_bounds__(256) void zero_kernel(float* __restrict__ p, int n) {
    const int i = blockIdx.x * 256 + threadIdx.x;
    if (i < n) p[i] = 0.0f;
}

// -------------------- gather + pooled-sum + abs-max (float4 / b128 traffic)
__global__ __launch_bounds__(256) void pool_absmax_kernel(
    const float* __restrict__ x, const int* __restrict__ idx,
    float* __restrict__ pooled, float* __restrict__ amaxv) {
    const int b   = blockIdx.z;
    const int tid = threadIdx.x;                  // owns cols [4*tid, 4*tid+3]
    const int k0  = blockIdx.x * 512;
    const float* xb = x + (size_t)b * S_ * D_;
    const int*   ib = idx + b * KSEL;
    float4 s  = {0.f, 0.f, 0.f, 0.f};
    float4 am = {0.f, 0.f, 0.f, 0.f};
    for (int k = k0; k < k0 + 512; ++k) {
        const int row = ib[k];                    // uniform -> scalar load
        const float4 v = ((const float4*)(xb + (size_t)row * D_))[tid];
        s.x += v.x; s.y += v.y; s.z += v.z; s.w += v.w;
        am.x = fmaxf(am.x, fabsf(v.x)); am.y = fmaxf(am.y, fabsf(v.y));
        am.z = fmaxf(am.z, fabsf(v.z)); am.w = fmaxf(am.w, fabsf(v.w));
    }
    float* pd = pooled + b * D_ + tid * 4;
    unsigned int* ad = (unsigned int*)(amaxv + b * D_ + tid * 4);
    atomicAdd(pd + 0, s.x); atomicAdd(pd + 1, s.y);
    atomicAdd(pd + 2, s.z); atomicAdd(pd + 3, s.w);
    atomicMax(ad + 0, __float_as_uint(am.x)); atomicMax(ad + 1, __float_as_uint(am.y));
    atomicMax(ad + 2, __float_as_uint(am.z)); atomicMax(ad + 3, __float_as_uint(am.w));
}

// -------------------------- quant MLP -> argmax(logits) -> 2^-bits per batch
__global__ __launch_bounds__(256) void bits_kernel(
    const float* __restrict__ pooled, const float* __restrict__ wq1,
    const float* __restrict__ bq1, const float* __restrict__ wq2,
    const float* __restrict__ bq2, float* __restrict__ pw) {
    __shared__ float pm[D_];
    __shared__ float hq[Q_];
    __shared__ float lg[8];
    const int b = blockIdx.x, tid = threadIdx.x;
    for (int i = tid; i < D_; i += 256) pm[i] = pooled[b * D_ + i] * (1.0f / KSEL);
    __syncthreads();
    float acc = bq1[tid];
    for (int d = 0; d < D_; ++d) acc += pm[d] * wq1[d * Q_ + tid];
    hq[tid] = fmaxf(acc, 0.f);
    __syncthreads();
    if (tid < 8) {
        float l = bq2[tid];
        for (int q = 0; q < Q_; ++q) l += hq[q] * wq2[q * 8 + tid];
        lg[tid] = l;
    }
    __syncthreads();
    if (tid == 0) {
        int arg = 0; float best = lg[0];
        for (int o = 1; o < 8; ++o) if (lg[o] > best) { best = lg[o]; arg = o; }
        pw[b] = exp2f(-(float)(arg + 1));          // 2^-num_bits
    }
}

// ------------------------------------- gather + quantize output (float4 IO)
__global__ __launch_bounds__(256) void quant_kernel(
    const float* __restrict__ x, const int* __restrict__ idx,
    const float* __restrict__ amaxv, const float* __restrict__ pw,
    float* __restrict__ out) {
    const int b   = blockIdx.y;
    const int tid = threadIdx.x;                   // owns cols [4*tid, 4*tid+3]
    const float p = pw[b];
    const float4 a = ((const float4*)(amaxv + b * D_))[tid];
    const float4 st = {a.x * p, a.y * p, a.z * p, a.w * p};
    const float* xb = x + (size_t)b * S_ * D_;
    float* ob = out + (size_t)b * KSEL * D_;
    for (int r = 0; r < 8; ++r) {
        const int k   = blockIdx.x * 8 + r;
        const int row = idx[b * KSEL + k];
        const float4 v = ((const float4*)(xb + (size_t)row * D_))[tid];
        float4 q;
        q.x = (st.x > 0.f) ? rintf(v.x / st.x) * st.x : 0.f;
        q.y = (st.y > 0.f) ? rintf(v.y / st.y) * st.y : 0.f;
        q.z = (st.z > 0.f) ? rintf(v.z / st.z) * st.z : 0.f;
        q.w = (st.w > 0.f) ? rintf(v.w / st.w) * st.w : 0.f;
        ((float4*)(ob + (size_t)k * D_))[tid] = q;
    }
}

// ---------------------------------------------------------------------------
extern "C" void kernel_launch(void* const* d_in, const int* in_sizes, int n_in,
                              void* d_out, int out_size, void* d_ws, size_t ws_size,
                              hipStream_t stream) {
    const float* x    = (const float*)d_in[0];
    const float* w1   = (const float*)d_in[1];
    const float* b1   = (const float*)d_in[2];
    const float* ln_g = (const float*)d_in[3];
    const float* ln_b = (const float*)d_in[4];
    const float* w2   = (const float*)d_in[5];
    const float* b2   = (const float*)d_in[6];
    const float* wq1  = (const float*)d_in[7];
    const float* bq1  = (const float*)d_in[8];
    const float* wq2  = (const float*)d_in[9];
    const float* bq2  = (const float*)d_in[10];
    float* out = (float*)d_out;

    // workspace layout
    char* ws = (char*)d_ws;
    _Float16* w1p    = (_Float16*)ws;                     ws += (size_t)D_ * H_ * 2;   // 1 MB
    float*    scores = (float*)ws;                        ws += (size_t)B_ * S_ * 4;   // 128 KB
    int*      idx    = (int*)ws;                          ws += (size_t)B_ * KSEL * 4; // 64 KB
    float*    pooled = (float*)ws;                        ws += (size_t)B_ * D_ * 4;   // 16 KB
    float*    amaxv  = (float*)ws;                        ws += (size_t)B_ * D_ * 4;   // 16 KB
    float*    pw     = (float*)ws;                        ws += 64;

    // 1) w1 -> fragment-major f16
    pack_w1_kernel<<<(D_ * H_) / 256, 256, 0, stream>>>(w1, w1p);
    // 2) importance scores (WMMA GEMM + LN + ReLU + proj + sigmoid)
    scorer_kernel<<<(B_ * S_) / 32, 256, 0, stream>>>(x, w1p, b1, ln_g, ln_b, w2, b2, scores);
    // 3) exact stable top-k per batch
    rank_kernel<<<dim3(S_ / 256, B_), 256, 0, stream>>>(scores, idx);
    // 4) zero accumulators, then pooled-sum + abs-max over selected rows
    zero_kernel<<<(2 * B_ * D_ + 255) / 256, 256, 0, stream>>>(pooled, 2 * B_ * D_);
    pool_absmax_kernel<<<dim3(KSEL / 512, 1, B_), 256, 0, stream>>>(x, idx, pooled, amaxv);
    // 5) bit-width decision per batch
    bits_kernel<<<B_, 256, 0, stream>>>(pooled, wq1, bq1, wq2, bq2, pw);
    // 6) gather + quantize to output
    quant_kernel<<<dim3(KSEL / 8, B_), 256, 0, stream>>>(x, idx, amaxv, pw, out);
}
